// MultiHeadAttention_23974507446481
// MI455X (gfx1250) — compile-verified, matrix operand order checked by reading, not run on hardware
//
#include <hip/hip_runtime.h>
#include <stdint.h>

#define B_  2
#define S_  2048
#define D_  1024
#define H_  16
#define DK_ 64

typedef unsigned short u16t;
typedef __attribute__((ext_vector_type(16))) __bf16 v16bf;
typedef __attribute__((ext_vector_type(8)))  float  v8f;
typedef __attribute__((ext_vector_type(4)))  int    v4i;

#if defined(__AMDGCN__) && __has_builtin(__builtin_amdgcn_global_load_async_to_lds_b128)
#define USE_ASYNC_LDS 1
#else
#define USE_ASYNC_LDS 0
#endif

union Frag16 { v16bf v; uint4 q[2]; };
union FragS  { v16bf v; u16t s[16]; };
union Acc8   { v8f v; float f[8]; };

// Load a 16-bit A/B WMMA fragment from two 16-byte-aligned runs of 8 bf16.
// CDNA5 layout: lanes 0-15 hold K = {0..7, 16..23}, lanes 16-31 K = {8..15, 24..31}
// (caller bakes the +8 half-offset into p0/p1).
__device__ __forceinline__ v16bf ld_frag(const u16t* p0, const u16t* p1) {
  Frag16 fr;
  fr.q[0] = *(const uint4*)p0;
  fr.q[1] = *(const uint4*)p1;
  return fr.v;
}

__device__ __forceinline__ u16t f2bf(float f) {
  uint32_t u = __float_as_uint(f);
  uint32_t r = u + 0x7fffu + ((u >> 16) & 1u);  // round-to-nearest-even
  return (u16t)(r >> 16);
}

// ---------------------------------------------------------------------------
// Kernel 0: fp32 -> bf16 conversion
// ---------------------------------------------------------------------------
__global__ void cvt_f32_to_bf16(const float* __restrict__ src,
                                u16t* __restrict__ dst, int n) {
  int i = blockIdx.x * blockDim.x + threadIdx.x;
  if (i < n) dst[i] = f2bf(src[i]);
}

// ---------------------------------------------------------------------------
// GEMM: out[m,n] = sum_k X[m,k] * W[n,k] + bias[n]
// X: [M x K] bf16 row-major, W: [N x K] bf16 row-major (i.e. we compute X·Wᵀ).
// Each wave owns a 64(M) x 64(N) tile = 16 f32 accumulators; 16 WMMAs per
// 16 global_load_b128 in the K loop (4x fragment reuse each way).
// ---------------------------------------------------------------------------
template<int F32OUT>
__global__ __launch_bounds__(256) void gemm_bf16_wmma(
    const u16t* __restrict__ X, const u16t* __restrict__ W,
    const float* __restrict__ bias,
    u16t* __restrict__ outB, float* __restrict__ outF,
    int M, int N, int K)
{
  const int lane  = threadIdx.x & 31;
  const int wave  = threadIdx.x >> 5;
  const int gtile = blockIdx.x * 8 + wave;
  const int tilesN = N / 64;
  const int m0 = (gtile / tilesN) * 64;
  const int n0 = (gtile % tilesN) * 64;
  const int laneM = lane & 15;
  const int hoff  = (lane < 16) ? 0 : 8;

  const u16t* pa[4];
  #pragma unroll
  for (int s = 0; s < 4; ++s) pa[s] = X + (size_t)(m0 + s * 16 + laneM) * K;
  const u16t* pb[4];
  #pragma unroll
  for (int j = 0; j < 4; ++j) pb[j] = W + (size_t)(n0 + j * 16 + laneM) * K;

  Acc8 acc[4][4];
  #pragma unroll
  for (int s = 0; s < 4; ++s)
    #pragma unroll
    for (int j = 0; j < 4; ++j)
      #pragma unroll
      for (int r = 0; r < 8; ++r) acc[s][j].f[r] = 0.0f;

  for (int k0 = 0; k0 < K; k0 += 32) {
    v16bf a[4], b[4];
    #pragma unroll
    for (int s = 0; s < 4; ++s)
      a[s] = ld_frag(pa[s] + k0 + hoff, pa[s] + k0 + 16 + hoff);
    #pragma unroll
    for (int j = 0; j < 4; ++j)
      b[j] = ld_frag(pb[j] + k0 + hoff, pb[j] + k0 + 16 + hoff);
    #pragma unroll
    for (int s = 0; s < 4; ++s)
      #pragma unroll
      for (int j = 0; j < 4; ++j)
        acc[s][j].v = __builtin_amdgcn_wmma_f32_16x16x32_bf16(
            false, a[s], false, b[j], (short)0, acc[s][j].v, false, false);
  }

  #pragma unroll
  for (int j = 0; j < 4; ++j) {
    const int col = n0 + j * 16 + laneM;
    const float bv = bias[col];
    #pragma unroll
    for (int s = 0; s < 4; ++s) {
      #pragma unroll
      for (int r = 0; r < 8; ++r) {
        const int m = m0 + s * 16 + ((lane < 16) ? r : (8 + r));
        const float o = acc[s][j].f[r] + bv;
        if (F32OUT) outF[(size_t)m * N + col] = o;
        else        outB[(size_t)m * N + col] = f2bf(o);
      }
    }
  }
}

// ---------------------------------------------------------------------------
// Flash attention (causal), bf16 WMMA, fp32 accumulation & online softmax.
// Grid: (B*H, S/64). Block: 128 (4 waves). Wave w handles q rows
// [qblk*64 + w*16, +16). kv processed in 32-wide chunks.
// V chunk is staged into LDS with GLOBAL_LOAD_ASYNC_TO_LDS_B128 (ASYNCcnt)
// when available; B fragments are gathered transposed on the read side.
// ---------------------------------------------------------------------------
__global__ __launch_bounds__(128) void flash_attn_bf16(
    const u16t* __restrict__ q, const u16t* __restrict__ k,
    const u16t* __restrict__ v, u16t* __restrict__ ctx)
{
#if USE_ASYNC_LDS
  __shared__ alignas(16) u16t vrow[32][64];    // V chunk, row-major [n][dd]
#else
  __shared__ alignas(16) u16t vt[64][40];      // V chunk transposed: vt[dd][n]
#endif
  __shared__ alignas(16) u16t pl[4][16][40];   // per-wave P tile (C->A relayout)

  const int lane  = threadIdx.x & 31;
  const int wave  = threadIdx.x >> 5;
  const int b     = blockIdx.x / H_;
  const int h     = blockIdx.x % H_;
  const int qblk  = blockIdx.y;
  const int laneM = lane & 15;
  const int hoff  = (lane < 16) ? 0 : 8;
  const int rowHalf = (lane < 16) ? 0 : 8;
  const size_t headOff = (size_t)h * DK_;

  // q A-fragments (16 rows x 64 d), kept in registers for the whole kernel
  const int qrow = qblk * 64 + wave * 16 + laneM;
  const u16t* qp = q + ((size_t)(b * S_ + qrow) * D_ + headOff);
  const v16bf aq0 = ld_frag(qp + hoff,      qp + 16 + hoff);
  const v16bf aq1 = ld_frag(qp + 32 + hoff, qp + 48 + hoff);

  float rmax[8], rsum[8];
  Acc8 o[4];
  #pragma unroll
  for (int r = 0; r < 8; ++r) { rmax[r] = -3.0e38f; rsum[r] = 0.0f; }
  #pragma unroll
  for (int f = 0; f < 4; ++f)
    #pragma unroll
    for (int r = 0; r < 8; ++r) o[f].f[r] = 0.0f;

  const int nChunks = qblk * 2 + 2;            // causal: kv up to qblk*64+63
  for (int c = 0; c < nChunks; ++c) {
    const int n0 = c * 32;

    __syncthreads();                           // prior LDS readers done
#if USE_ASYNC_LDS
    {   // async DMA stage of V[n0..n0+31, h*64..h*64+63], row-major, 2x16B/thread
      const int seg0 = threadIdx.x * 2;
      #pragma unroll
      for (int t = 0; t < 2; ++t) {
        const int e  = (seg0 + t) * 8;         // bf16 element offset in 32x64 tile
        const int rn = e >> 6;                 // kv row in chunk
        const int dd = e & 63;                 // feature within head
        const u16t* gp = v + ((size_t)(b * S_ + n0 + rn) * D_ + headOff + dd);
        __builtin_amdgcn_global_load_async_to_lds_b128(
            (__attribute__((address_space(1))) v4i*)gp,
            (__attribute__((address_space(3))) v4i*)&vrow[rn][dd],
            0, 0);
      }
#if __has_builtin(__builtin_amdgcn_s_wait_asynccnt)
      __builtin_amdgcn_s_wait_asynccnt(0);
#else
      asm volatile("s_wait_asynccnt 0x0" ::: "memory");
#endif
    }
#else
    {   // cooperative transposed stage of V[n0..n0+31, h*64..h*64+63]
      const int rn = threadIdx.x >> 2;         // 0..31 (kv row in chunk)
      const int cg = threadIdx.x & 3;          // 0..3  (16-col group)
      const u16t* vp = v + ((size_t)(b * S_ + n0 + rn) * D_ + headOff + cg * 16);
      uint4 d0 = *(const uint4*)(vp);
      uint4 d1 = *(const uint4*)(vp + 8);
      u16t tmp[16];
      *(uint4*)(&tmp[0]) = d0;
      *(uint4*)(&tmp[8]) = d1;
      #pragma unroll
      for (int e = 0; e < 16; ++e) vt[cg * 16 + e][rn] = tmp[e];
    }
#endif
    __syncthreads();

    // prefetch next chunk's K and V rows (global_prefetch_b8)
    if (c + 1 < nChunks) {
      const int pr = n0 + 32 + laneM + ((lane < 16) ? 0 : 16);
      const size_t prOff = (size_t)(b * S_ + pr) * D_ + headOff;
      __builtin_prefetch(k + prOff, 0, 3);
      __builtin_prefetch(v + prOff, 0, 3);
    }

    // S-tile 16x32: two 16x16 accumulators, K fragments straight from global
    Acc8 sacc[2];
    #pragma unroll
    for (int j = 0; j < 2; ++j) {
      #pragma unroll
      for (int r = 0; r < 8; ++r) sacc[j].f[r] = 0.0f;
      const int krow = n0 + j * 16 + laneM;
      const u16t* kp = k + ((size_t)(b * S_ + krow) * D_ + headOff);
      v16bf bk0 = ld_frag(kp + hoff,      kp + 16 + hoff);
      v16bf bk1 = ld_frag(kp + 32 + hoff, kp + 48 + hoff);
      sacc[j].v = __builtin_amdgcn_wmma_f32_16x16x32_bf16(
          false, aq0, false, bk0, (short)0, sacc[j].v, false, false);
      sacc[j].v = __builtin_amdgcn_wmma_f32_16x16x32_bf16(
          false, aq1, false, bk1, (short)0, sacc[j].v, false, false);
    }

    // scale 1/sqrt(64), causal mask, online softmax in C-layout
    float p0[8], p1[8], cmax[8];
    #pragma unroll
    for (int r = 0; r < 8; ++r) {
      const int mg = qblk * 64 + wave * 16 + rowHalf + r;
      float s0 = (n0 + laneM      <= mg) ? sacc[0].f[r] * 0.125f : -1.0e9f;
      float s1 = (n0 + 16 + laneM <= mg) ? sacc[1].f[r] * 0.125f : -1.0e9f;
      p0[r] = s0; p1[r] = s1;
      cmax[r] = fmaxf(s0, s1);
    }
    #pragma unroll
    for (int r = 0; r < 8; ++r) {
      #pragma unroll
      for (int mk = 8; mk >= 1; mk >>= 1)
        cmax[r] = fmaxf(cmax[r], __shfl_xor(cmax[r], mk, 16));
      const float nm   = fmaxf(rmax[r], cmax[r]);
      const float corr = __expf(rmax[r] - nm);
      rmax[r] = nm;
      p0[r] = __expf(p0[r] - nm);
      p1[r] = __expf(p1[r] - nm);
      float rs = p0[r] + p1[r];
      #pragma unroll
      for (int mk = 8; mk >= 1; mk >>= 1)
        rs += __shfl_xor(rs, mk, 16);
      rsum[r] = rsum[r] * corr + rs;
      #pragma unroll
      for (int f = 0; f < 4; ++f) o[f].f[r] *= corr;
    }

    // P: C-layout -> per-wave LDS -> A-fragment (same-wave DS ops are in order)
    #pragma unroll
    for (int r = 0; r < 8; ++r) {
      const int pm = rowHalf + r;
      pl[wave][pm][laneM]      = f2bf(p0[r]);
      pl[wave][pm][16 + laneM] = f2bf(p1[r]);
    }
    const u16t* pp = &pl[wave][laneM][0];
    v16bf ap = ld_frag(pp + hoff, pp + 16 + hoff);

    // ctx += P(16x32) x V(32x64)
    #pragma unroll
    for (int f = 0; f < 4; ++f) {
      v16bf bvf;
#if USE_ASYNC_LDS
      // transposed read-gather from row-major LDS: lane = N = dd, elems along K = n
      FragS fb;
      const int dd = f * 16 + laneM;
      #pragma unroll
      for (int i = 0; i < 16; ++i) {
        const int n = ((i < 8) ? i : (i + 8)) + rowHalf;
        fb.s[i] = vrow[n][dd];
      }
      bvf = fb.v;
#else
      const u16t* pv = &vt[f * 16 + laneM][0];
      bvf = ld_frag(pv + hoff, pv + 16 + hoff);
#endif
      o[f].v = __builtin_amdgcn_wmma_f32_16x16x32_bf16(
          false, ap, false, bvf, (short)0, o[f].v, false, false);
    }
  }

  // normalize, store ctx as [B, S, D] bf16 (heads already interleaved in D)
  #pragma unroll
  for (int f = 0; f < 4; ++f) {
    #pragma unroll
    for (int r = 0; r < 8; ++r) {
      const int m  = qblk * 64 + wave * 16 + rowHalf + r;
      const int dd = f * 16 + laneM;
      ctx[(size_t)(b * S_ + m) * D_ + headOff + dd] = f2bf(o[f].f[r] / rsum[r]);
    }
  }
}

// ---------------------------------------------------------------------------
extern "C" void kernel_launch(void* const* d_in, const int* in_sizes, int n_in,
                              void* d_out, int out_size, void* d_ws, size_t ws_size,
                              hipStream_t stream) {
  (void)in_sizes; (void)n_in; (void)out_size; (void)ws_size;
  const float* Q  = (const float*)d_in[0];
  const float* K  = (const float*)d_in[1];
  const float* V  = (const float*)d_in[2];
  // d_in[3] = causal mask (tril) — deterministic, computed analytically in-kernel
  const float* Wq = (const float*)d_in[4];
  const float* bq = (const float*)d_in[5];
  const float* Wk = (const float*)d_in[6];
  const float* bk = (const float*)d_in[7];
  const float* Wv = (const float*)d_in[8];
  const float* bv = (const float*)d_in[9];
  const float* Wo = (const float*)d_in[10];
  const float* bo = (const float*)d_in[11];
  float* out = (float*)d_out;

  const size_t nQ = (size_t)B_ * S_ * D_;   // 4,194,304
  const size_t nW = (size_t)D_ * D_;        // 1,048,576
  u16t* w   = (u16t*)d_ws;                  // 64 MB total bf16 scratch
  u16t* Qb  = w;
  u16t* Kb  = w + 1 * nQ;
  u16t* Vb  = w + 2 * nQ;
  u16t* qp  = w + 3 * nQ;
  u16t* kp  = w + 4 * nQ;
  u16t* vp  = w + 5 * nQ;
  u16t* cx  = w + 6 * nQ;
  u16t* Wqb = w + 7 * nQ;
  u16t* Wkb = Wqb + 1 * nW;
  u16t* Wvb = Wqb + 2 * nW;
  u16t* Wob = Wqb + 3 * nW;

  const int T = 256;
  cvt_f32_to_bf16<<<(int)(nQ / T), T, 0, stream>>>(Q,  Qb,  (int)nQ);
  cvt_f32_to_bf16<<<(int)(nQ / T), T, 0, stream>>>(K,  Kb,  (int)nQ);
  cvt_f32_to_bf16<<<(int)(nQ / T), T, 0, stream>>>(V,  Vb,  (int)nQ);
  cvt_f32_to_bf16<<<(int)(nW / T), T, 0, stream>>>(Wq, Wqb, (int)nW);
  cvt_f32_to_bf16<<<(int)(nW / T), T, 0, stream>>>(Wk, Wkb, (int)nW);
  cvt_f32_to_bf16<<<(int)(nW / T), T, 0, stream>>>(Wv, Wvb, (int)nW);
  cvt_f32_to_bf16<<<(int)(nW / T), T, 0, stream>>>(Wo, Wob, (int)nW);

  const int M = B_ * S_;                    // 4096
  // tiles = (M/64)*(D/64) = 1024 wave-tiles / 8 waves per block = 128 blocks
  gemm_bf16_wmma<0><<<128, 256, 0, stream>>>(Qb, Wqb, bq, qp, nullptr, M, D_, D_);
  gemm_bf16_wmma<0><<<128, 256, 0, stream>>>(Kb, Wkb, bk, kp, nullptr, M, D_, D_);
  gemm_bf16_wmma<0><<<128, 256, 0, stream>>>(Vb, Wvb, bv, vp, nullptr, M, D_, D_);

  flash_attn_bf16<<<dim3(B_ * H_, S_ / 64), 128, 0, stream>>>(qp, kp, vp, cx);

  gemm_bf16_wmma<1><<<128, 256, 0, stream>>>(cx, Wob, bo, nullptr, out, M, D_, D_);
}
